// NeuralODE_41686952575792
// MI455X (gfx1250) — compile-verified, matrix operand order checked by reading, not run on hardware
//
#include <hip/hip_runtime.h>
#include <hip/hip_bf16.h>
#include <math.h>

typedef __attribute__((ext_vector_type(16))) _Float16 v16h;
typedef __attribute__((ext_vector_type(8)))  _Float16 v8h;
typedef __attribute__((ext_vector_type(8)))  float    v8f;

// ---------------- problem constants ----------------
constexpr int T_STEPS = 64;
constexpr int WV      = 4;     // waves per workgroup
constexpr int ROWS    = 16;    // batch rows per wave (WMMA N with swapped operands)

// Tsit5 coefficients
constexpr float C2f=0.161f, C3f=0.327f, C4f=0.9f, C5f=0.9800255409045097f;
constexpr float A21f=0.161f;
constexpr float A31f=-0.008480655492356989f, A32f=0.335480655492357f;
constexpr float A41f=2.8971530571054935f, A42f=-6.359448489975075f, A43f=4.3622954328695815f;
constexpr float A51f=5.325864828439257f, A52f=-11.748883564062828f, A53f=7.4955393428898365f, A54f=-0.09249506636175525f;
constexpr float A61f=5.86145544294642f, A62f=-12.92096931784711f, A63f=8.159367898576159f, A64f=-0.071584973281401f, A65f=-0.028269050394068383f;
constexpr float B1f=0.09646076681806523f, B2f=0.01f, B3f=0.4798896504144996f, B4f=1.379008574103742f, B5f=-3.290069515436081f, B6f=2.324710524099774f;

__constant__ float TSA[6][5] = {
  { 0.f,   0.f,   0.f,   0.f,   0.f  },
  { A21f,  0.f,   0.f,   0.f,   0.f  },
  { A31f,  A32f,  0.f,   0.f,   0.f  },
  { A41f,  A42f,  A43f,  0.f,   0.f  },
  { A51f,  A52f,  A53f,  A54f,  0.f  },
  { A61f,  A62f,  A63f,  A64f,  A65f },
};
__constant__ float TSC[6] = { 0.f, C2f, C3f, C4f, C5f, 1.f };
__constant__ float TSB[6] = { B1f, B2f, B3f, B4f, B5f, B6f };

// ---------------- LDS layout ----------------
// f16 weight region (offsets in halves), row-major [Npad][Kpad], zero padded
constexpr int W_ENC1 = 0;                    // 32x32
constexpr int W_ENC2 = W_ENC1 + 32*32;       // 32x32
constexpr int W_ENC3 = W_ENC2 + 32*32;       // 32x32
constexpr int W_ENC4 = W_ENC3 + 32*32;       // 16x32
constexpr int W_CTX1 = W_ENC4 + 16*32;       // 64x32  (49x3 real)
constexpr int W_CTX2 = W_CTX1 + 64*32;       // 96x64  (96x49 real)
constexpr int W_DAT1 = W_CTX2 + 96*64;       // 32x32  (32x8 real)
constexpr int W_DAT2 = W_DAT1 + 32*32;       // 32x32
constexpr int W_M1   = W_DAT2 + 32*32;       // 32x64 (M1..M3 contiguous)
constexpr int W_M2   = W_M1   + 32*64;
constexpr int W_M3   = W_M2   + 32*64;
constexpr int W_M4   = W_M3   + 32*64;       // 16x32  (8x32 real)
constexpr int W_TOT  = W_M4   + 16*32;       // 20480 halves
constexpr int WBYTES = W_TOT * 2;            // 40960 B

// f32 bias region (offsets in floats), zero padded
constexpr int B_ENC1=0,  B_ENC2=32, B_ENC3=64, B_ENC4=96;
constexpr int B_CTX1=112, B_CTX2=176;
constexpr int B_DAT1=272, B_DAT2=304;
constexpr int B_M1=336,  B_M2=368,  B_M3=400, B_M4=432;
constexpr int B_TOT=448;
constexpr int BIAS_BYTE  = WBYTES;                  // 40960
constexpr int WAVE_BYTE0 = BIAS_BYTE + B_TOT*4;     // 42752

// per-wave scratch (byte offsets within wave slab)
constexpr int PW_DELTAS = 0;                 // 16*64*2 f32 = 8192
constexpr int PW_BUFX   = 8192;              // 16*96 f16 = 3072  (also xs staging)
constexpr int PW_BUFY   = 8192 + 3072;
constexpr int PW_BUFC   = 8192 + 6144;
constexpr int PW_Z      = 8192 + 9216;       // 128 f32
constexpr int PW_K      = PW_Z + 512;        // 6 * 128 f32
constexpr int PW_ZT     = PW_K + 6*512;      // 128 f32
constexpr int PW_ENCO   = PW_ZT + 512;       // 16*16 f32
constexpr int PW_SZ     = PW_ENCO + 1024;    // 22528
constexpr int LDS_BYTES = WAVE_BYTE0 + WV * PW_SZ;   // 132864

// ---------------- helpers ----------------
__device__ __forceinline__ void wave_sync() {
  asm volatile("s_wait_dscnt 0" ::: "memory");
  __builtin_amdgcn_wave_barrier();
}
__device__ __forceinline__ void async_wait() {
  asm volatile("s_wait_asynccnt 0" ::: "memory");
  __builtin_amdgcn_wave_barrier();
}

// fast activations
__device__ __forceinline__ float swish_f(float v, float beta) {
  return v * __builtin_amdgcn_rcpf(1.f + __expf(-beta * v));
}
__device__ __forceinline__ float softplus_f(float v) {
  return (v > 20.f) ? v : __logf(1.f + __expf(v));
}

// One lane's WMMA operand fragment from row-major f16 data.
// lanes 0-15: K = {k0..k0+7, k0+16..k0+23}; lanes 16-31: {k0+8..,k0+24..}
__device__ __forceinline__ v16h ld_frag(const _Float16* p) {
  v8h lo = *(const v8h*)p;
  v8h hi = *(const v8h*)(p + 16);
  v16h r;
#pragma unroll
  for (int i = 0; i < 8; ++i) { r[i] = lo[i]; r[i + 8] = hi[i]; }
  return r;
}
__device__ __forceinline__ void copy8(_Float16* dst, const _Float16* src) {
  *(v8h*)dst = *(const v8h*)src;
}

// D = W @ act^T  (A-operand = weight tile, B-operand = activation rows).
// D layout: lane = batch row m (l16), VGPR r = output feature; this lane's
// 8 features are consecutive: n = nt*16 + half*8 + r  -> one v8h store.
// ACT: 0 = linear, 1 = swish(beta), 2 = softplus
template<int ACT, int KT, int NT>
__device__ __forceinline__ void gemm_layer(const _Float16* act, int strideA,
                                           const _Float16* W, const float* bias,
                                           int NrealF, float beta,
                                           _Float16* outH, int strideH,
                                           float* outF, int strideF, int lane)
{
  const int half = lane >> 4;
  const int l16  = lane & 15;    // batch row m
#pragma unroll
  for (int nt = 0; nt < NT; ++nt) {
    v8f c = {};
#pragma unroll
    for (int kt = 0; kt < KT; ++kt) {
      v16h a = ld_frag(W + (nt * 16 + l16) * (KT * 32) + kt * 32 + half * 8);
      v16h b = ld_frag(act + l16 * strideA + kt * 32 + half * 8);
      c = __builtin_amdgcn_wmma_f32_16x16x32_f16(false, a, false, b, (short)0, c,
                                                 false, false);
    }
    const int nb = nt * 16 + half * 8;     // first of this lane's 8 features
    float4 bs0 = *(const float4*)(bias + nb);
    float4 bs1 = *(const float4*)(bias + nb + 4);
    float vv[8];
    vv[0]=c[0]+bs0.x; vv[1]=c[1]+bs0.y; vv[2]=c[2]+bs0.z; vv[3]=c[3]+bs0.w;
    vv[4]=c[4]+bs1.x; vv[5]=c[5]+bs1.y; vv[6]=c[6]+bs1.z; vv[7]=c[7]+bs1.w;
#pragma unroll
    for (int r = 0; r < 8; ++r) {
      if (ACT == 1)      vv[r] = swish_f(vv[r], beta);
      else if (ACT == 2) vv[r] = softplus_f(vv[r]);
    }
    if (outH) {
      v8h oh;
#pragma unroll
      for (int r = 0; r < 8; ++r) oh[r] = (_Float16)vv[r];
      *(v8h*)(outH + l16 * strideH + nb) = oh;
    }
    if (outF && nb < NrealF) {
      *(float4*)(outF + l16 * strideF + nb)     = make_float4(vv[0],vv[1],vv[2],vv[3]);
      *(float4*)(outF + l16 * strideF + nb + 4) = make_float4(vv[4],vv[5],vv[6],vv[7]);
    }
  }
  wave_sync();
}

struct Wv {
  _Float16 *bufX, *bufY, *bufC;
  float    *deltas;           // [16][64][2]
  _Float16 *wH;
  float    *bF;
  float betaCtx, betaDat, bm0, bm1, bm2;
  float t0, dtg, inv_dt;
  int   lane;
};

// kout[16*8] = vector field at time t for state zin[16*8]
__device__ __forceinline__ void vf(const Wv& C, float t,
                                   const float* zin, float* kout)
{
  const int lane = C.lane;
  // ---- ctx = interp(deltas, t) into bufX cols [0,32), 8-half chunks
  {
    float pos = (t - C.t0) * C.inv_dt;
    float fi  = floorf(pos);
    fi = fminf(fmaxf(fi, 0.f), (float)(T_STEPS - 2));
    int   ii   = (int)fi;
    float frac = pos - fi;
    for (int ch = lane; ch < 64; ch += 32) {
      int m = ch >> 2, q = ch & 3;
      v8h val = {};
      if (q == 0) {
        const float* dm = C.deltas + (m * T_STEPS + ii) * 2;
        float lo0 = dm[0], lo1 = dm[1], hi0 = dm[2], hi1 = dm[3];
        float lo2 = (ii == 0) ? 0.f : C.dtg;
        val[0] = (_Float16)(lo0 + frac * (hi0 - lo0));
        val[1] = (_Float16)(lo1 + frac * (hi1 - lo1));
        val[2] = (_Float16)(lo2 + frac * (C.dtg - lo2));
      }
      *(v8h*)(C.bufX + m * 96 + q * 8) = val;
    }
    wave_sync();
  }
  // ctx net: 3->49 swish, 49->96 linear
  gemm_layer<1, 1, 4>(C.bufX, 96, C.wH + W_CTX1, C.bF + B_CTX1, 0, C.betaCtx,
                      C.bufY, 96, nullptr, 0, lane);
  gemm_layer<0, 2, 6>(C.bufY, 96, C.wH + W_CTX2, C.bF + B_CTX2, 0, 0.f,
                      C.bufC, 96, nullptr, 0, lane);
  // dat input: z (8 wide, zero pad to 32) -- 8-half chunks
  for (int ch = lane; ch < 64; ch += 32) {
    int m = ch >> 2, q = ch & 3;
    v8h val = {};
    if (q == 0) {
      float4 z0 = *(const float4*)(zin + m * 8);
      float4 z1 = *(const float4*)(zin + m * 8 + 4);
      val[0]=(_Float16)z0.x; val[1]=(_Float16)z0.y; val[2]=(_Float16)z0.z; val[3]=(_Float16)z0.w;
      val[4]=(_Float16)z1.x; val[5]=(_Float16)z1.y; val[6]=(_Float16)z1.z; val[7]=(_Float16)z1.w;
    }
    *(v8h*)(C.bufX + m * 96 + q * 8) = val;
  }
  wave_sync();
  gemm_layer<1, 1, 2>(C.bufX, 96, C.wH + W_DAT1, C.bF + B_DAT1, 0, C.betaDat,
                      C.bufY, 96, nullptr, 0, lane);
  gemm_layer<0, 1, 2>(C.bufY, 96, C.wH + W_DAT2, C.bF + B_DAT2, 0, 0.f,
                      C.bufX, 96, nullptr, 0, lane);
  // main layers: y = swish([y, cp_l] @ W_l^T + b_l)
#pragma unroll
  for (int l = 0; l < 3; ++l) {
    const float beta_l = (l == 0) ? C.bm0 : (l == 1) ? C.bm1 : C.bm2;
    for (int ch = lane; ch < 64; ch += 32) {          // cp -> cols [32,64)
      int m = ch >> 2, q = ch & 3;
      copy8(C.bufX + m * 96 + 32 + q * 8, C.bufC + m * 96 + l * 32 + q * 8);
    }
    wave_sync();
    gemm_layer<1, 2, 2>(C.bufX, 96, C.wH + W_M1 + l * (32 * 64),
                        C.bF + B_M1 + l * 32, 0, beta_l,
                        C.bufY, 96, nullptr, 0, lane);
    if (l < 2) {
      for (int ch = lane; ch < 64; ch += 32) {        // y -> cols [0,32)
        int m = ch >> 2, q = ch & 3;
        copy8(C.bufX + m * 96 + q * 8, C.bufY + m * 96 + q * 8);
      }
      wave_sync();
    }
  }
  // output layer 32->8 (f32 to kout; only lanes with nb==0 store)
  gemm_layer<0, 1, 1>(C.bufY, 96, C.wH + W_M4, C.bF + B_M4, 8, 0.f,
                      nullptr, 0, kout, 8, lane);
}

__device__ __forceinline__ void load_w(_Float16* dst, const float* src,
                                       int N, int K, int Npad, int Kpad,
                                       int tid, int nthr) {
  for (int i = tid; i < Npad * Kpad; i += nthr) {
    int n = i / Kpad, kk = i % Kpad;
    float v = (n < N && kk < K) ? src[n * K + kk] : 0.f;
    dst[i] = (_Float16)v;
  }
}
__device__ __forceinline__ void load_b(float* dst, const float* src,
                                       int N, int Npad, int tid, int nthr) {
  for (int i = tid; i < Npad; i += nthr) dst[i] = (i < N) ? src[i] : 0.f;
}

struct P33 { const float* p[33]; };

__global__ __launch_bounds__(128, 1)
void node_kernel(P33 ps, float* __restrict__ out, int Btot)
{
  extern __shared__ __align__(16) char smem[];
  _Float16* wH = (_Float16*)smem;
  float*    bF = (float*)(smem + BIAS_BYTE);

  const int tid  = threadIdx.x;
  const int lane = tid & 31;
  const int wave = tid >> 5;

  // ---- cache all weights in LDS as padded f16, biases as padded f32
  load_w(wH + W_ENC1, ps.p[2],  32,  2, 32, 32, tid, 128);
  load_w(wH + W_ENC2, ps.p[4],  32, 32, 32, 32, tid, 128);
  load_w(wH + W_ENC3, ps.p[6],  32, 32, 32, 32, tid, 128);
  load_w(wH + W_ENC4, ps.p[8],  16, 32, 16, 32, tid, 128);
  load_w(wH + W_CTX1, ps.p[10], 49,  3, 64, 32, tid, 128);
  load_w(wH + W_CTX2, ps.p[12], 96, 49, 96, 64, tid, 128);
  load_w(wH + W_DAT1, ps.p[15], 32,  8, 32, 32, tid, 128);
  load_w(wH + W_DAT2, ps.p[17], 32, 32, 32, 32, tid, 128);
  load_w(wH + W_M1,   ps.p[20], 32, 64, 32, 64, tid, 128);
  load_w(wH + W_M2,   ps.p[22], 32, 64, 32, 64, tid, 128);
  load_w(wH + W_M3,   ps.p[24], 32, 64, 32, 64, tid, 128);
  load_w(wH + W_M4,   ps.p[26],  8, 32, 16, 32, tid, 128);
  load_b(bF + B_ENC1, ps.p[3],  32, 32, tid, 128);
  load_b(bF + B_ENC2, ps.p[5],  32, 32, tid, 128);
  load_b(bF + B_ENC3, ps.p[7],  32, 32, tid, 128);
  load_b(bF + B_ENC4, ps.p[9],  16, 16, tid, 128);
  load_b(bF + B_CTX1, ps.p[11], 49, 64, tid, 128);
  load_b(bF + B_CTX2, ps.p[13], 96, 96, tid, 128);
  load_b(bF + B_DAT1, ps.p[16], 32, 32, tid, 128);
  load_b(bF + B_DAT2, ps.p[18], 32, 32, tid, 128);
  load_b(bF + B_M1,   ps.p[21], 32, 32, tid, 128);
  load_b(bF + B_M2,   ps.p[23], 32, 32, tid, 128);
  load_b(bF + B_M3,   ps.p[25], 32, 32, tid, 128);
  load_b(bF + B_M4,   ps.p[27],  8, 16, tid, 128);
  __syncthreads();

  // ---- per-wave slab
  char* pw = smem + WAVE_BYTE0 + wave * PW_SZ;
  Wv C;
  C.deltas = (float*)(pw + PW_DELTAS);
  C.bufX   = (_Float16*)(pw + PW_BUFX);
  C.bufY   = (_Float16*)(pw + PW_BUFY);
  C.bufC   = (_Float16*)(pw + PW_BUFC);
  C.wH = wH;  C.bF = bF;  C.lane = lane;
  C.betaCtx = ps.p[14][0];
  C.betaDat = ps.p[19][0];
  C.bm0 = ps.p[28][0];  C.bm1 = ps.p[29][0];  C.bm2 = ps.p[30][0];
  const float* tev = ps.p[1];
  C.t0  = tev[0];
  C.dtg = tev[1] - tev[0];
  C.inv_dt = 1.f / C.dtg;

  float* z  = (float*)(pw + PW_Z);
  float* zt = (float*)(pw + PW_ZT);
  float* kb = (float*)(pw + PW_K);     // 6 x 128 floats
  float* enc = (float*)(pw + PW_ENCO);

  const float* xs = ps.p[0];
  const int b0 = (blockIdx.x * WV + wave) * ROWS;
  if (b0 >= Btot) return;

  const size_t XR = 0;
  const size_t ZL = (size_t)Btot * T_STEPS * 2;
  const size_t MU = ZL + (size_t)Btot * 8;
  const size_t LV = MU + (size_t)Btot * 8;

  // ---- async-stage this wave's xs tile (16 rows x 64 x 2 f32 = 8KB,
  // contiguous in memory) into bufX..bufC region via ASYNCcnt path
  {
    float* stage = (float*)(pw + PW_BUFX);
    const uint64_t gbase = (uint64_t)(uintptr_t)(xs + (size_t)b0 * T_STEPS * 2);
    const uint32_t lbase = (uint32_t)(uintptr_t)stage;
#pragma unroll 4
    for (int ch = lane; ch < 512; ch += 32) {       // 512 x 16B chunks
      uint32_t laddr = lbase + ch * 16;
      uint64_t gaddr = gbase + (uint64_t)ch * 16;
      asm volatile("global_load_async_to_lds_b128 %0, %1, off"
                   :: "v"(laddr), "v"(gaddr) : "memory");
    }
    async_wait();
    // deltas of [xs, t]: diff along time, prepend first (=> 0)
    for (int idx = lane; idx < ROWS * T_STEPS; idx += 32) {
      int m = idx / T_STEPS, j = idx % T_STEPS;
      const float* row = stage + (m * T_STEPS + j) * 2;
      float d0 = 0.f, d1 = 0.f;
      if (j > 0) { d0 = row[0] - row[-2]; d1 = row[1] - row[-1]; }
      C.deltas[(m * T_STEPS + j) * 2 + 0] = d0;
      C.deltas[(m * T_STEPS + j) * 2 + 1] = d1;
    }
    wave_sync();
    // encoder input xs[:,0,:] -> bufY cols[0,32)
    for (int ch = lane; ch < 64; ch += 32) {
      int m = ch >> 2, q = ch & 3;
      v8h val = {};
      if (q == 0) {
        val[0] = (_Float16)stage[m * T_STEPS * 2 + 0];
        val[1] = (_Float16)stage[m * T_STEPS * 2 + 1];
      }
      *(v8h*)(C.bufY + m * 96 + q * 8) = val;
    }
    wave_sync();
  }
  // ---- encoder
  gemm_layer<2, 1, 2>(C.bufY, 96, wH + W_ENC1, bF + B_ENC1, 0, 0.f, C.bufX, 96, nullptr, 0, lane);
  gemm_layer<2, 1, 2>(C.bufX, 96, wH + W_ENC2, bF + B_ENC2, 0, 0.f, C.bufY, 96, nullptr, 0, lane);
  gemm_layer<2, 1, 2>(C.bufY, 96, wH + W_ENC3, bF + B_ENC3, 0, 0.f, C.bufX, 96, nullptr, 0, lane);
  gemm_layer<0, 1, 1>(C.bufX, 96, wH + W_ENC4, bF + B_ENC4, 16, 0.f, nullptr, 0, enc, 16, lane);

  // mu -> z and to out; logvar to out (float4 per lane)
  {
    int m = lane >> 1, d0 = (lane & 1) * 4;
    float4 mu = *(const float4*)(enc + m * 16 + d0);
    float4 lv = *(const float4*)(enc + m * 16 + 8 + d0);
    *(float4*)(z + m * 8 + d0) = mu;
    *(float4*)(out + MU + (size_t)(b0 + m) * 8 + d0) = mu;
    *(float4*)(out + LV + (size_t)(b0 + m) * 8 + d0) = lv;
  }
  wave_sync();

  const float* decW = ps.p[31];
  const float* decb = ps.p[32];
  {
    int m = lane >> 1, n = lane & 1;
    float acc = decb[n];
#pragma unroll
    for (int kk = 0; kk < 8; ++kk) acc += z[m * 8 + kk] * decW[n * 8 + kk];
    out[XR + ((size_t)(b0 + m) * T_STEPS + 0) * 2 + n] = acc;
  }

  // ---- 126 Tsit5 substeps; single vf inline instance (stage loop)
  const float hs = C.dtg * 0.5f;
  const int i0 = lane * 4;          // this lane's float4 chunk of [16][8]
#pragma unroll 1
  for (int ss = 0; ss < 2 * (T_STEPS - 1); ++ss) {
    const int it = ss >> 1;
    const int s  = ss & 1;
    const float tb = tev[it] + s * hs;
#pragma unroll 1
    for (int st = 0; st < 6; ++st) {
      const float* zin = z;
      if (st > 0) {
        float4 zc = *(const float4*)(z + i0);
        float ax = 0.f, ay = 0.f, az = 0.f, aw = 0.f;
        for (int j = 0; j < st; ++j) {
          float aj = TSA[st][j];
          float4 kc = *(const float4*)(kb + j * 128 + i0);
          ax += aj * kc.x; ay += aj * kc.y; az += aj * kc.z; aw += aj * kc.w;
        }
        *(float4*)(zt + i0) = make_float4(zc.x + hs * ax, zc.y + hs * ay,
                                          zc.z + hs * az, zc.w + hs * aw);
        wave_sync();
        zin = zt;
      }
      vf(C, tb + TSC[st] * hs, zin, kb + st * 128);
    }
    // z += hs * sum(Bi ki)
    {
      float4 zc = *(const float4*)(z + i0);
      float ax = 0.f, ay = 0.f, az = 0.f, aw = 0.f;
#pragma unroll
      for (int j = 0; j < 6; ++j) {
        float bj = TSB[j];
        float4 kc = *(const float4*)(kb + j * 128 + i0);
        ax += bj * kc.x; ay += bj * kc.y; az += bj * kc.z; aw += bj * kc.w;
      }
      *(float4*)(z + i0) = make_float4(zc.x + hs * ax, zc.y + hs * ay,
                                       zc.z + hs * az, zc.w + hs * aw);
      wave_sync();
    }
    // decoder for zs[:, it+1] after the second substep of the interval
    if (s == 1) {
      int m = lane >> 1, n = lane & 1;
      float acc = decb[n];
#pragma unroll
      for (int kk = 0; kk < 8; ++kk) acc += z[m * 8 + kk] * decW[n * 8 + kk];
      out[XR + ((size_t)(b0 + m) * T_STEPS + (it + 1)) * 2 + n] = acc;
    }
  }
  // z_last (float4 per lane)
  {
    int m = lane >> 1, d0 = (lane & 1) * 4;
    float4 zl = *(const float4*)(z + m * 8 + d0);
    *(float4*)(out + ZL + (size_t)(b0 + m) * 8 + d0) = zl;
  }
}

extern "C" void kernel_launch(void* const* d_in, const int* in_sizes, int n_in,
                              void* d_out, int out_size, void* d_ws, size_t ws_size,
                              hipStream_t stream)
{
  (void)d_ws; (void)ws_size; (void)out_size;
  P33 ps;
  for (int i = 0; i < 33 && i < n_in; ++i) ps.p[i] = (const float*)d_in[i];
  const int B = in_sizes[0] / (T_STEPS * 2);          // xs is (B, 64, 2)
  const int blocks = (B + ROWS * WV - 1) / (ROWS * WV);
  (void)hipFuncSetAttribute((const void*)node_kernel,
                            hipFuncAttributeMaxDynamicSharedMemorySize,
                            LDS_BYTES);
  hipLaunchKernelGGL(node_kernel, dim3(blocks), dim3(WV * 32), LDS_BYTES, stream,
                     ps, (float*)d_out, B);
}
